// HierarchicalVQ_1735166787742
// MI455X (gfx1250) — matrix operand order repaired and checked by reading, not compiled, not run on hardware
//
#include <hip/hip_runtime.h>
#include <math.h>

#define DIMC 512
#define KCB  4096
#define NB   32
#define HW2  32
#define NPIXEL (NB*HW2*HW2)      // 32768 pixels
#define NELEM  (NPIXEL*DIMC)     // 16777216 activations
#define COMMIT_POS 16777216
#define IDX_POS    16777217
#define NIDX 43680

#if defined(__has_builtin)
#if __has_builtin(__builtin_amdgcn_sched_barrier)
#define SCHED_FENCE() __builtin_amdgcn_sched_barrier(0)
#endif
#endif
#ifndef SCHED_FENCE
#define SCHED_FENCE()
#endif

typedef __attribute__((ext_vector_type(16))) __bf16 v16bf;
typedef __attribute__((ext_vector_type(8)))  float  v8f;
typedef __attribute__((ext_vector_type(8)))  unsigned short ushort8;

union Frag {
  v16bf   v;
  ushort8 h[2];
};

__device__ __forceinline__ unsigned short f2bf(float f) {
  union { float f; unsigned int u; } cv; cv.f = f;
  unsigned int u = cv.u;
  unsigned int r = 0x7FFFu + ((u >> 16) & 1u);   // round-to-nearest-even
  return (unsigned short)((u + r) >> 16);
}

// ---- init: x NCHW -> residual NHWC, zero recon + commit accumulators ----
__global__ void init_kernel(const float* __restrict__ x, float* __restrict__ res,
                            float* __restrict__ recon, float* __restrict__ cacc) {
  if (blockIdx.x == 0 && threadIdx.x < 8) cacc[threadIdx.x] = 0.0f;
  for (int i = blockIdx.x * blockDim.x + threadIdx.x; i < NELEM;
       i += gridDim.x * blockDim.x) {
    int c = i & 511, xx = (i >> 9) & 31, y = (i >> 14) & 31, b = i >> 19;
    res[i]   = x[(b << 19) + (c << 10) + (y << 5) + xx];
    recon[i] = 0.0f;
  }
}

// ---- codebook -> bf16 + per-code squared norms ----
__global__ void prep_codebook(const float* __restrict__ cb,
                              unsigned short* __restrict__ cbf,
                              float* __restrict__ cnorm) {
  __shared__ float red[256];
  int k = blockIdx.x, t = threadIdx.x;
  float s = 0.f;
  for (int c = t; c < DIMC; c += 256) {
    float v = cb[k * DIMC + c];
    cbf[k * DIMC + c] = f2bf(v);
    s += v * v;
  }
  red[t] = s; __syncthreads();
  for (int off = 128; off > 0; off >>= 1) {
    if (t < off) red[t] += red[t + off];
    __syncthreads();
  }
  if (t == 0) cnorm[k] = red[0];
}

// ---- phi_w (O,I,3,3) -> bf16 layout [kw][co][ci] (ci innermost for B-frags) ----
__global__ void prep_weights(const float* __restrict__ w, unsigned short* __restrict__ wt) {
  const int total = DIMC * DIMC * 9;
  for (int o = blockIdx.x * blockDim.x + threadIdx.x; o < total;
       o += gridDim.x * blockDim.x) {
    int ci = o & 511, t = o >> 9, co = t & 511, kw = t >> 9;
    wt[o] = f2bf(w[(co * DIMC + ci) * 9 + kw]);
  }
}

// ---- adaptive avg pool: residual NHWC -> tokens (N x 512) f32 + bf16 ----
__global__ void pool_kernel(const float* __restrict__ res, float* __restrict__ tokf,
                            unsigned short* __restrict__ tokb, int s) {
  int N = NB * s * s;
  int blk = HW2 / s;
  float inv = 1.0f / (float)(blk * blk);
  int total = N * DIMC;
  for (int i = blockIdx.x * blockDim.x + threadIdx.x; i < total;
       i += gridDim.x * blockDim.x) {
    int c = i & 511, n = i >> 9;
    int j = n % s, t = n / s, ii = t % s, b = t / s;
    int y0 = ii * blk, x0 = j * blk;
    float acc = 0.f;
    for (int dy = 0; dy < blk; ++dy)
      for (int dx = 0; dx < blk; ++dx)
        acc += res[((size_t)((b * 32 + (y0 + dy)) * 32 + (x0 + dx))) * DIMC + c];
    float m = acc * inv;
    tokf[i] = m;
    tokb[i] = f2bf(m);
  }
}

// ---- VQ argmin: one wave = 16 tokens; A panel in VGPRs, fenced batch B loads ----
__global__ void __launch_bounds__(32, 1)
vq_kernel(const unsigned short* __restrict__ tokb,
          const unsigned short* __restrict__ cbf,
          const float* __restrict__ cnorm,
          int* __restrict__ idx_out) {
  __shared__ float sc[16][17];
  int lane = threadIdx.x;
  int col = lane & 15, half = lane >> 4;
  int n0 = blockIdx.x * 16;
  const unsigned short* arow = tokb + (size_t)(n0 + col) * DIMC;

  // A panel: 16 tokens x 512 K, resident in 128 VGPRs for the whole kernel
  Frag areg[16];
  #pragma unroll
  for (int t = 0; t < 16; ++t) {
    areg[t].h[0] = *(const ushort8*)(arow + t * 32 + 16 * half);       // K 16h..
    areg[t].h[1] = *(const ushort8*)(arow + t * 32 + 16 * half + 8);   // K 16h+8..
  }

  float bestv = 3.4e38f; int besti = 0;
  for (int k0 = 0; k0 < KCB; k0 += 16) {
    if (k0 + 16 < KCB)
      __builtin_prefetch(cbf + (size_t)(k0 + 16 + col) * DIMC, 0, 1);
    const unsigned short* brow = cbf + (size_t)(k0 + col) * DIMC + 16 * half;
    v8f acc = {0.f,0.f,0.f,0.f,0.f,0.f,0.f,0.f};

    // Batch 1: 8 B fragments; sched fence pins all 16 b128 loads before the
    // first WMMA -> distinct regs, in-order completion, partial loadcnt waits.
    Frag b[8];
    #pragma unroll
    for (int p = 0; p < 8; ++p) {
      b[p].h[0] = *(const ushort8*)(brow + p * 32);                    // K 16h..
      b[p].h[1] = *(const ushort8*)(brow + p * 32 + 8);
    }
    SCHED_FENCE();
    #pragma unroll
    for (int t = 0; t < 8; ++t)
      acc = __builtin_amdgcn_wmma_f32_16x16x32_bf16(false, areg[t].v, false,
                                                    b[t].v, (short)0, acc,
                                                    false, false);
    SCHED_FENCE();
    // Batch 2: chunks 8..15
    #pragma unroll
    for (int p = 0; p < 8; ++p) {
      b[p].h[0] = *(const ushort8*)(brow + (p + 8) * 32);
      b[p].h[1] = *(const ushort8*)(brow + (p + 8) * 32 + 8);
    }
    SCHED_FENCE();
    #pragma unroll
    for (int t = 0; t < 8; ++t)
      acc = __builtin_amdgcn_wmma_f32_16x16x32_bf16(false, areg[8 + t].v, false,
                                                    b[t].v, (short)0, acc,
                                                    false, false);

    #pragma unroll
    for (int v = 0; v < 8; ++v) sc[v + 8 * half][col] = acc[v];
    __syncthreads();
    if (lane < 16) {
      #pragma unroll
      for (int j = 0; j < 16; ++j) {
        float d = cnorm[k0 + j] - 2.0f * sc[lane][j];
        if (d < bestv) { bestv = d; besti = k0 + j; }
      }
    }
    __syncthreads();
  }
  if (lane < 16) idx_out[n0 + lane] = besti;
}

// ---- commit MSE partial sums ----
__global__ void commit_kernel(const float* __restrict__ tokf, const float* __restrict__ cb,
                              const int* __restrict__ idx, float* __restrict__ acc_slot,
                              int N) {
  __shared__ float red[256];
  int total = N * DIMC;
  float s = 0.f;
  for (int i = blockIdx.x * blockDim.x + threadIdx.x; i < total;
       i += gridDim.x * blockDim.x) {
    int c = i & 511, n = i >> 9;
    float d = cb[(size_t)idx[n] * DIMC + c] - tokf[i];
    s += d * d;
  }
  int t = threadIdx.x;
  red[t] = s; __syncthreads();
  for (int off = 128; off > 0; off >>= 1) {
    if (t < off) red[t] += red[t + off];
    __syncthreads();
  }
  if (t == 0) atomicAdd(acc_slot, red[0]);
}

// ---- gather codebook rows + bilinear upsample (half-pixel) -> q NHWC f32+bf16 ----
__global__ void upsample_kernel(const float* __restrict__ cb, const int* __restrict__ idx,
                                float* __restrict__ qf, unsigned short* __restrict__ qb,
                                int s) {
  int bid = blockIdx.x;
  int xx = bid & 31, y = (bid >> 5) & 31, b = bid >> 10;
  float scale = (float)s / 32.0f;
  float cy = (y + 0.5f) * scale - 0.5f;
  float cx = (xx + 0.5f) * scale - 0.5f;
  int i0 = (int)floorf(cy); float fy = cy - (float)i0;
  int j0 = (int)floorf(cx); float fx = cx - (float)j0;
  int i1 = i0 + 1, j1 = j0 + 1;
  i0 = min(s - 1, max(0, i0)); i1 = min(s - 1, max(0, i1));
  j0 = min(s - 1, max(0, j0)); j1 = min(s - 1, max(0, j1));
  const int base = b * s * s;
  const float* r00 = cb + (size_t)idx[base + i0 * s + j0] * DIMC;
  const float* r01 = cb + (size_t)idx[base + i0 * s + j1] * DIMC;
  const float* r10 = cb + (size_t)idx[base + i1 * s + j0] * DIMC;
  const float* r11 = cb + (size_t)idx[base + i1 * s + j1] * DIMC;
  float w00 = (1.f - fy) * (1.f - fx), w01 = (1.f - fy) * fx;
  float w10 = fy * (1.f - fx),         w11 = fy * fx;
  size_t o = (size_t)bid * DIMC;
  for (int c = threadIdx.x; c < DIMC; c += blockDim.x) {
    float v = w00 * r00[c] + w01 * r01[c] + w10 * r10[c] + w11 * r11[c];
    qf[o + c] = v;
    qb[o + c] = f2bf(v);
  }
}

// ---- phi: implicit-GEMM conv3x3, 16 px x 64 co per wave, fenced double-buffer ----
__global__ void __launch_bounds__(32, 1)
conv_kernel(const unsigned short* __restrict__ qb, const float* __restrict__ qf,
            const unsigned short* __restrict__ wt, const float* __restrict__ bias,
            float* __restrict__ phi) {
  int bid = blockIdx.x;
  int cg  = (bid & 7) << 6;  bid >>= 3;   // 64-channel output group
  int x0  = (bid & 1) << 4;  bid >>= 1;   // 16-pixel x tile
  int y   = bid & 31;        int b = bid >> 5;
  int lane = threadIdx.x;
  int col = lane & 15, half = lane >> 4;
  const ushort8 zero8 = {0,0,0,0,0,0,0,0};
  v8f acc[4];
  #pragma unroll
  for (int nt = 0; nt < 4; ++nt) acc[nt] = (v8f){0.f,0.f,0.f,0.f,0.f,0.f,0.f,0.f};

  #pragma unroll
  for (int dy = 0; dy < 3; ++dy) {
    int yy = y + dy - 1;
    bool rok = (yy >= 0) && (yy < 32);
    #pragma unroll
    for (int dx = 0; dx < 3; ++dx) {
      int xp = x0 + col + dx - 1;                 // A row = pixel per lane
      bool ok = rok && (xp >= 0) && (xp < 32);
      const unsigned short* arow = qb + (size_t)((b * 32 + yy) * 32 + xp) * DIMC;
      const unsigned short* wb =
          wt + ((size_t)(dy * 3 + dx) * DIMC + (cg + col)) * DIMC + 16 * half;

      Frag a[2], w[2][4];
      // prologue: chunk 0
      if (ok) {
        a[0].h[0] = *(const ushort8*)(arow + 16 * half);
        a[0].h[1] = *(const ushort8*)(arow + 16 * half + 8);
      } else { a[0].h[0] = zero8; a[0].h[1] = zero8; }
      #pragma unroll
      for (int nt = 0; nt < 4; ++nt) {
        const unsigned short* brow = wb + (size_t)(nt * 16) * DIMC;
        w[0][nt].h[0] = *(const ushort8*)(brow);
        w[0][nt].h[1] = *(const ushort8*)(brow + 8);
      }
      #pragma unroll
      for (int i = 0; i < 16; ++i) {
        int cur = i & 1, nxt = cur ^ 1;
        if (i < 15) {                             // fetch chunk i+1 early
          int cb = (i + 1) * 32;
          if (ok) {
            a[nxt].h[0] = *(const ushort8*)(arow + cb + 16 * half);
            a[nxt].h[1] = *(const ushort8*)(arow + cb + 16 * half + 8);
          } else { a[nxt].h[0] = zero8; a[nxt].h[1] = zero8; }
          #pragma unroll
          for (int nt = 0; nt < 4; ++nt) {
            const unsigned short* brow = wb + (size_t)(nt * 16) * DIMC;
            w[nxt][nt].h[0] = *(const ushort8*)(brow + cb);
            w[nxt][nt].h[1] = *(const ushort8*)(brow + cb + 8);
          }
        }
        SCHED_FENCE();                            // pin prefetch batch above WMMAs
        #pragma unroll
        for (int nt = 0; nt < 4; ++nt)            // A reused across 4 co-tiles
          acc[nt] = __builtin_amdgcn_wmma_f32_16x16x32_bf16(false, a[cur].v, false,
                                                            w[cur][nt].v, (short)0,
                                                            acc[nt], false, false);
        SCHED_FENCE();
      }
    }
  }
  size_t rowbase = (size_t)((b * 32 + y) * 32) * DIMC;
  #pragma unroll
  for (int nt = 0; nt < 4; ++nt) {
    int co = cg + nt * 16 + col;
    float bv = bias[co];
    #pragma unroll
    for (int v = 0; v < 8; ++v) {
      int px = x0 + (v + 8 * half);               // D: m = vgpr + 8*half
      size_t o = rowbase + (size_t)px * DIMC + co;
      phi[o] = 0.5f * qf[o] + 0.5f * (acc[nt][v] + bv); // (1-r)q + r*conv
    }
  }
}

// ---- recon += phi ; residual -= phi ----
__global__ void update_kernel(const float* __restrict__ phi, float* __restrict__ recon,
                              float* __restrict__ res) {
  for (int i = blockIdx.x * blockDim.x + threadIdx.x; i < NELEM;
       i += gridDim.x * blockDim.x) {
    float p = phi[i];
    recon[i] += p;
    res[i]   -= p;
  }
}

// ---- recon NHWC -> NCHW into d_out ----
__global__ void finalize_kernel(const float* __restrict__ recon, float* __restrict__ out) {
  for (int i = blockIdx.x * blockDim.x + threadIdx.x; i < NELEM;
       i += gridDim.x * blockDim.x) {
    int xx = i & 31, yy = (i >> 5) & 31, c = (i >> 10) & 511, b = i >> 19;
    out[i] = recon[(size_t)((b * 32 + yy) * 32 + xx) * DIMC + c];
  }
}

__global__ void write_commit(const float* __restrict__ cacc, float* __restrict__ out) {
  const float ninv[6] = {1.f/(32*512), 1.f/(128*512), 1.f/(512*512),
                         1.f/(2048*512), 1.f/(8192*512), 1.f/(32768*512)};
  float m = 0.f;
  for (int i = 0; i < 6; ++i) m += cacc[i] * ninv[i];
  out[COMMIT_POS] = m * (1.0f / 6.0f);
}

__global__ void write_idx(const int* __restrict__ idx, float* __restrict__ out) {
  int i = blockIdx.x * blockDim.x + threadIdx.x;
  if (i < NIDX) out[IDX_POS + i] = (float)idx[i];
}

extern "C" void kernel_launch(void* const* d_in, const int* in_sizes, int n_in,
                              void* d_out, int out_size, void* d_ws, size_t ws_size,
                              hipStream_t stream) {
  const float* x  = (const float*)d_in[0];
  const float* cb = (const float*)d_in[1];
  const float* pw = (const float*)d_in[2];
  const float* pb = (const float*)d_in[3];
  float* out = (float*)d_out;
  char* ws = (char*)d_ws;

  const size_t SZ = (size_t)NELEM * 4;                    // 64 MiB per f32 plane
  float* res   = (float*)(ws);
  float* recon = (float*)(ws + SZ);
  float* qf    = (float*)(ws + 2 * SZ);
  float* phi   = (float*)(ws + 3 * SZ);
  float* tokf  = (float*)(ws + 4 * SZ);
  unsigned short* qb   = (unsigned short*)(ws + 5 * SZ);
  unsigned short* tokb = (unsigned short*)(ws + 5 * SZ + SZ / 2);
  unsigned short* cbf  = (unsigned short*)(ws + 6 * SZ);
  size_t off = 6 * SZ + (size_t)KCB * DIMC * 2;
  float* cnorm = (float*)(ws + off);          off += (size_t)KCB * 4;
  unsigned short* wt = (unsigned short*)(ws + off); off += (size_t)9 * DIMC * DIMC * 2;
  int* idx_all = (int*)(ws + off);            off += (size_t)NIDX * 4;
  float* cacc  = (float*)(ws + off);          off += 64;

  init_kernel<<<8192, 256, 0, stream>>>(x, res, recon, cacc);
  prep_codebook<<<KCB, 256, 0, stream>>>(cb, cbf, cnorm);
  prep_weights<<<4608, 256, 0, stream>>>(pw, wt);

  static const int scales[6] = {1, 2, 4, 8, 16, 32};
  static const int ioff[6]   = {0, 32, 160, 672, 2720, 10912};
  for (int si = 0; si < 6; ++si) {
    int s = scales[si];
    int N = NB * s * s;
    int tot = N * DIMC;
    pool_kernel<<<(tot + 255) / 256, 256, 0, stream>>>(res, tokf, tokb, s);
    int* idx = idx_all + ioff[si];
    vq_kernel<<<N / 16, 32, 0, stream>>>(tokb, cbf, cnorm, idx);
    commit_kernel<<<(tot + 255) / 256, 256, 0, stream>>>(tokf, cb, idx, cacc + si, N);
    upsample_kernel<<<NPIXEL, 256, 0, stream>>>(cb, idx, qf, qb, s);
    conv_kernel<<<16384, 32, 0, stream>>>(qb, qf, wt, pb, phi);
    update_kernel<<<8192, 256, 0, stream>>>(phi, recon, res);
  }

  finalize_kernel<<<8192, 256, 0, stream>>>(recon, out);
  write_commit<<<1, 1, 0, stream>>>(cacc, out);
  write_idx<<<(NIDX + 255) / 256, 256, 0, stream>>>(idx_all, out);
}
